// Model_23751169147163
// MI455X (gfx1250) — matrix operand order repaired and checked by reading, not compiled, not run on hardware
//
#include <hip/hip_runtime.h>

// ---------------------------------------------------------------------------
// Sizes from the reference
// ---------------------------------------------------------------------------
#define ENT_DIM   256
#define HDIM      256
#define EBATCH    2048
#define SEQ       32
#define NPAIR     1024
#define REL_SIZE  500
#define REL_DIM   256

typedef __attribute__((ext_vector_type(16))) __bf16 v16bf;
typedef __attribute__((ext_vector_type(8)))  float  v8f;

struct __attribute__((aligned(16))) U4 { unsigned x, y, z, w; };
struct __attribute__((aligned(16))) U8 { U4 a, b; };

// Build a 16-element bf16 WMMA fragment from two 16-byte chunks.
__device__ __forceinline__ v16bf frag2(const __bf16* p0, const __bf16* p1) {
    U8 u;
    u.a = *reinterpret_cast<const U4*>(p0);
    u.b = *reinterpret_cast<const U4*>(p1);
    return __builtin_bit_cast(v16bf, u);
}

__device__ __forceinline__ v8f wmma_bf16(v16bf a, v16bf b, v8f c) {
    return __builtin_amdgcn_wmma_f32_16x16x32_bf16(
        /*neg_a=*/false, a, /*neg_b=*/false, b,
        /*c_mod=*/(short)0, c, /*reuse_a=*/false, /*reuse_b=*/false);
}

__device__ __forceinline__ float fsigmoid(float x) {
    return 1.0f / (1.0f + __expf(-x));
}
__device__ __forceinline__ float ftanh(float x) {
    float e = __expf(2.0f * x);
    return (e - 1.0f) / (e + 1.0f);
}

// ---------------------------------------------------------------------------
// 0) init: zero scalar output and BN partial-sum buffers
// ---------------------------------------------------------------------------
__global__ void init_kernel(float* out, float* sum, float* sumsq) {
    int t = threadIdx.x;
    if (t == 0) out[0] = 0.0f;
    sum[t]   = 0.0f;
    sumsq[t] = 0.0f;
}

// ---------------------------------------------------------------------------
// 1) fp32 -> bf16 weight conversion
// ---------------------------------------------------------------------------
__global__ void cvt_bf16_kernel(const float* __restrict__ src,
                                __bf16* __restrict__ dst, int n) {
    int i = blockIdx.x * blockDim.x + threadIdx.x;
    if (i < n) dst[i] = (__bf16)src[i];
}

// ---------------------------------------------------------------------------
// 2) LSTM (one direction per workgroup half of grid)
//    grid = 256 blocks: blk[0..127] fwd E-tiles, blk[128..255] bwd E-tiles
//    block = 256 threads = 8 waves; wave w owns hidden cols [32w, 32w+32)
//    of EVERY gate, so the cell state c lives entirely in registers.
//    Loop order: one A fragment live at a time; B (weights) streamed from
//    L2 every step (an asm memory clobber stops LLVM from hoisting the
//    loop-invariant weight fragments into scratch spills).
// ---------------------------------------------------------------------------
#define XPITCH 264   // bf16 elements per LDS row (256 + 8 pad -> conflict-free)

__global__ __launch_bounds__(256) void lstm_kernel(
    const int*    __restrict__ neighbors,   // [E][S]
    const float*  __restrict__ ent_embed,   // [ENT_SIZE+1][256]
    const __bf16* __restrict__ wih_f, const __bf16* __restrict__ whh_f,
    const __bf16* __restrict__ wih_b, const __bf16* __restrict__ whh_b,
    const float*  __restrict__ b_f,   const float*  __restrict__ b_b,
    __bf16*       __restrict__ hout)        // [2][EBATCH][HDIM]
{
    __shared__ __bf16 xbuf[16 * XPITCH];
    __shared__ __bf16 hbuf[16 * XPITCH];

    const int blk   = blockIdx.x;
    const int dir   = blk >> 7;
    const int e0    = (blk & 127) * 16;
    const __bf16* Wih  = dir ? wih_b : wih_f;
    const __bf16* Whh  = dir ? whh_b : whh_f;
    const float*  bias = dir ? b_b   : b_f;

    const int t     = threadIdx.x;
    const int lane  = t & 31;
    const int wv    = t >> 5;       // wave 0..7
    const int ln16  = lane & 15;
    const int khalf = lane >> 4;    // 0 or 1

    // tile -> gate column base: tile = g*2 + half, col = g*HDIM + wv*32 + half*16
    // per-lane weight row pointers (B[k][n] = W[col+ln16][k], contiguous per lane)

    // zero initial hidden state in LDS
    for (int i = t; i < 16 * XPITCH; i += 256) hbuf[i] = (__bf16)0.0f;

    v8f c0 = {0.f,0.f,0.f,0.f,0.f,0.f,0.f,0.f};
    v8f c1 = {0.f,0.f,0.f,0.f,0.f,0.f,0.f,0.f};

    __syncthreads();

    for (int s = 0; s < SEQ; ++s) {
        // Stop LLVM from hoisting the (loop-invariant) weight loads out of
        // the step loop: re-stream them from L2 instead of spilling.
        asm volatile("" ::: "memory");

        // ---- gather x_t = ent_embed[neighbors[e, step]] -> LDS (bf16) ----
        {
            const int m    = t >> 4;            // row 0..15
            const int cseg = (t & 15) * 16;     // 16 floats per thread
            const int step = dir ? (SEQ - 1 - s) : s;
            const int nb   = neighbors[(e0 + m) * SEQ + step];
            const float* src = ent_embed + (size_t)nb * ENT_DIM + cseg;
            __bf16* dst = xbuf + m * XPITCH + cseg;
            #pragma unroll
            for (int i = 0; i < 16; i += 4) {
                float4 v = *reinterpret_cast<const float4*>(src + i);
                dst[i + 0] = (__bf16)v.x;
                dst[i + 1] = (__bf16)v.y;
                dst[i + 2] = (__bf16)v.z;
                dst[i + 3] = (__bf16)v.w;
            }
        }
        __syncthreads();

        // ---- init accumulators with bias ----
        v8f acc[8];
        #pragma unroll
        for (int g = 0; g < 4; ++g) {
            #pragma unroll
            for (int half = 0; half < 2; ++half) {
                const int col = g * HDIM + wv * 32 + half * 16;
                const float bv = bias[col + ln16];
                v8f a = {bv, bv, bv, bv, bv, bv, bv, bv};
                acc[g * 2 + half] = a;
            }
        }

        // ---- G += X W_ih^T : one A fragment live, 8 tiles per kc ----
        {
            const __bf16* xrow = xbuf + ln16 * XPITCH;
            #pragma unroll
            for (int kc = 0; kc < 8; ++kc) {
                const int k0 = kc * 32 + khalf * 8;
                v16bf a = frag2(xrow + k0, xrow + k0 + 16);
                #pragma unroll
                for (int g = 0; g < 4; ++g) {
                    #pragma unroll
                    for (int half = 0; half < 2; ++half) {
                        const int col = g * HDIM + wv * 32 + half * 16 + ln16;
                        const __bf16* p = Wih + (size_t)col * ENT_DIM
                                              + kc * 32 + khalf * 8;
                        acc[g * 2 + half] =
                            wmma_bf16(a, frag2(p, p + 16), acc[g * 2 + half]);
                    }
                }
            }
        }

        // ---- G += H W_hh^T ----
        {
            const __bf16* hrow = hbuf + ln16 * XPITCH;
            #pragma unroll
            for (int kc = 0; kc < 8; ++kc) {
                const int k0 = kc * 32 + khalf * 8;
                v16bf a = frag2(hrow + k0, hrow + k0 + 16);
                #pragma unroll
                for (int g = 0; g < 4; ++g) {
                    #pragma unroll
                    for (int half = 0; half < 2; ++half) {
                        const int col = g * HDIM + wv * 32 + half * 16 + ln16;
                        const __bf16* p = Whh + (size_t)col * HDIM
                                              + kc * 32 + khalf * 8;
                        acc[g * 2 + half] =
                            wmma_bf16(a, frag2(p, p + 16), acc[g * 2 + half]);
                    }
                }
            }
        }

        __syncthreads();   // all waves done reading hbuf

        // ---- gate nonlinearity + cell update (registers), write new h ----
        #pragma unroll
        for (int half = 0; half < 2; ++half) {
            v8f i_ = acc[0 * 2 + half];
            v8f f_ = acc[1 * 2 + half];
            v8f g_ = acc[2 * 2 + half];
            v8f o_ = acc[3 * 2 + half];
            v8f& c = half ? c1 : c0;
            const int nb = wv * 32 + half * 16 + ln16;
            #pragma unroll
            for (int j = 0; j < 8; ++j) {
                float ci = fsigmoid(i_[j]);
                float cf = fsigmoid(f_[j]);
                float co = fsigmoid(o_[j]);
                float cg = ftanh(g_[j]);
                float cn = cf * c[j] + ci * cg;
                c[j] = cn;
                float hv = co * ftanh(cn);
                const int m = j + khalf * 8;   // C/D layout row
                hbuf[m * XPITCH + nb] = (__bf16)hv;
            }
        }
        __syncthreads();
    }

    // ---- write final hidden state (bf16) to global ----
    {
        __bf16* dst = hout + ((size_t)dir * EBATCH + e0) * HDIM;
        for (int i = t; i < 16 * HDIM; i += 256) {
            const int m  = i >> 8;
            const int cc = i & 255;
            dst[m * HDIM + cc] = hbuf[m * XPITCH + cc];
        }
    }
}

// ---------------------------------------------------------------------------
// 3) lin = concat(h_f, h_b) @ W_lin^T + b_lin    ([2048,512]x[512,256], WMMA)
//    grid = 256 blocks x 8 waves = 2048 waves = 128 M-tiles x 16 N-tiles
// ---------------------------------------------------------------------------
__global__ __launch_bounds__(256) void linear_kernel(
    const __bf16* __restrict__ hf, const __bf16* __restrict__ hb,
    const __bf16* __restrict__ wlin,   // [256][512] bf16 row-major
    const float*  __restrict__ blin,
    float*        __restrict__ lin)    // [2048][256]
{
    const int t     = threadIdx.x;
    const int lane  = t & 31;
    const int wv    = t >> 5;
    const int ln16  = lane & 15;
    const int khalf = lane >> 4;
    const int wid   = blockIdx.x * 8 + wv;   // 0..2047
    const int mtile = wid >> 4;
    const int ntile = wid & 15;
    const int m0    = mtile * 16;
    const int ncol  = ntile * 16 + ln16;

    const float bv = blin[ncol];
    v8f acc = {bv, bv, bv, bv, bv, bv, bv, bv};

    const __bf16* wrow  = wlin + (size_t)ncol * 512;
    const __bf16* arowf = hf + (size_t)(m0 + ln16) * HDIM;
    const __bf16* arowb = hb + (size_t)(m0 + ln16) * HDIM;

    #pragma unroll
    for (int kc = 0; kc < 16; ++kc) {
        const int kb = kc * 32;
        const __bf16* ar = (kb < 256) ? (arowf + kb) : (arowb + (kb - 256));
        const int k0 = khalf * 8;
        v16bf a = frag2(ar + k0, ar + k0 + 16);
        const __bf16* p = wrow + kb + khalf * 8;
        acc = wmma_bf16(a, frag2(p, p + 16), acc);
    }

    #pragma unroll
    for (int j = 0; j < 8; ++j)
        lin[(size_t)(m0 + j + khalf * 8) * 256 + ntile * 16 + ln16] = acc[j];
}

// ---------------------------------------------------------------------------
// 4) BatchNorm stats: per-column sum / sumsq over the 2048-row batch
// ---------------------------------------------------------------------------
__global__ void stats_kernel(const float* __restrict__ lin,
                             float* __restrict__ sum, float* __restrict__ sumsq) {
    const int c = threadIdx.x;   // column 0..255
    const int b = blockIdx.x;    // 32 row groups
    float s = 0.0f, s2 = 0.0f;
    for (int r = b; r < EBATCH; r += 32) {
        float v = lin[(size_t)r * 256 + c];
        s  += v;
        s2 += v * v;
    }
    atomicAdd(&sum[c],   s);
    atomicAdd(&sumsq[c], s2);
}

// ---------------------------------------------------------------------------
// 5) ctx = tanh(gamma * (lin - mu) * rsqrt(var + eps) + beta)
// ---------------------------------------------------------------------------
__global__ void ctx_kernel(const float* __restrict__ lin,
                           const float* __restrict__ sum, const float* __restrict__ sumsq,
                           const float* __restrict__ gamma, const float* __restrict__ beta,
                           float* __restrict__ ctx) {
    const int i = blockIdx.x * 256 + threadIdx.x;
    const int c = i & 255;
    const float inv = 1.0f / (float)EBATCH;
    const float mu  = sum[c] * inv;
    const float var = sumsq[c] * inv - mu * mu;
    const float x   = (lin[i] - mu) * rsqrtf(var + 1e-5f);
    ctx[i] = ftanh(gamma[c] * x + beta[c]);
}

// ---------------------------------------------------------------------------
// 6) scores: one workgroup per (pos|neg, n); batched GEMV over trans,
//    L2 norm, atomic accumulate sum(pos + relu(1 - neg)) into out[0]
// ---------------------------------------------------------------------------
__global__ __launch_bounds__(256) void score_kernel(
    const float* __restrict__ trans,     // [1000][256][256]
    const float* __restrict__ rel,       // [500][256]
    const float* __restrict__ ctx,       // [2048][256]
    const int* __restrict__ ph, const int* __restrict__ pt, const int* __restrict__ pr,
    const int* __restrict__ nh, const int* __restrict__ nt, const int* __restrict__ nr,
    float* __restrict__ out)
{
    __shared__ float ch[256];
    __shared__ float ct_[256];
    __shared__ float red[256];

    const int b   = blockIdx.x;
    const bool neg = (b >= NPAIR);
    const int n   = neg ? b - NPAIR : b;
    const int hi  = neg ? nh[n] : ph[n];
    const int ti  = neg ? nt[n] : pt[n];
    const int ri  = neg ? nr[n] : pr[n];
    const int t   = threadIdx.x;

    ch[t]  = ctx[(size_t)hi * 256 + t];
    ct_[t] = ctx[(size_t)ti * 256 + t];
    __syncthreads();

    const float* Wh = trans + ((size_t)ri)              * 65536 + (size_t)t * 256;
    const float* Wt = trans + ((size_t)(ri + REL_SIZE)) * 65536 + (size_t)t * 256;
    float dh = 0.0f, dt = 0.0f;
    #pragma unroll 4
    for (int k = 0; k < 256; k += 4) {
        float4 wh4 = *reinterpret_cast<const float4*>(Wh + k);
        float4 wt4 = *reinterpret_cast<const float4*>(Wt + k);
        dh += wh4.x * ch[k]  + wh4.y * ch[k + 1]  + wh4.z * ch[k + 2]  + wh4.w * ch[k + 3];
        dt += wt4.x * ct_[k] + wt4.y * ct_[k + 1] + wt4.z * ct_[k + 2] + wt4.w * ct_[k + 3];
    }
    const float d = dh - dt + rel[(size_t)ri * 256 + t];
    red[t] = d * d;
    __syncthreads();

    for (int off = 128; off > 0; off >>= 1) {
        if (t < off) red[t] += red[t + off];
        __syncthreads();
    }
    if (t == 0) {
        const float s = sqrtf(red[0]);
        atomicAdd(out, neg ? fmaxf(0.0f, 1.0f - s) : s);
    }
}

// ---------------------------------------------------------------------------
// launch
// ---------------------------------------------------------------------------
extern "C" void kernel_launch(void* const* d_in, const int* in_sizes, int n_in,
                              void* d_out, int out_size, void* d_ws, size_t ws_size,
                              hipStream_t stream) {
    const int*   neighbors = (const int*)  d_in[0];
    const int*   pos_h     = (const int*)  d_in[1];
    const int*   pos_t     = (const int*)  d_in[2];
    const int*   pos_r     = (const int*)  d_in[3];
    const int*   neg_h     = (const int*)  d_in[4];
    const int*   neg_t     = (const int*)  d_in[5];
    const int*   neg_r     = (const int*)  d_in[6];
    const float* ent_embed = (const float*)d_in[7];
    const float* rel_embed = (const float*)d_in[8];
    const float* trans     = (const float*)d_in[9];
    const float* W_ih_f    = (const float*)d_in[10];
    const float* W_hh_f    = (const float*)d_in[11];
    const float* W_ih_b    = (const float*)d_in[12];
    const float* W_hh_b    = (const float*)d_in[13];
    const float* b_f       = (const float*)d_in[14];
    const float* b_b       = (const float*)d_in[15];
    const float* W_lin     = (const float*)d_in[16];
    const float* b_lin     = (const float*)d_in[17];
    const float* gamma     = (const float*)d_in[18];
    const float* beta      = (const float*)d_in[19];
    float* out = (float*)d_out;

    // workspace layout (bytes) --------------------------------------------
    char* ws = (char*)d_ws;
    __bf16* wb_ih_f = (__bf16*)(ws + 0);         // 1024*256*2 = 524288
    __bf16* wb_hh_f = (__bf16*)(ws + 524288);
    __bf16* wb_ih_b = (__bf16*)(ws + 1048576);
    __bf16* wb_hh_b = (__bf16*)(ws + 1572864);
    __bf16* wb_lin  = (__bf16*)(ws + 2097152);   // 256*512*2 = 262144
    __bf16* hout    = (__bf16*)(ws + 2359296);   // 2*2048*256*2 = 2097152
    float*  lin     = (float*) (ws + 4456448);   // 2048*256*4 = 2097152
    float*  bn_sum  = (float*) (ws + 6553600);   // 256*4
    float*  bn_sq   = (float*) (ws + 6554624);   // 256*4
    float*  ctx     = (float*) (ws + 6555648);   // 2048*256*4 = 2097152
    // total ~8.25 MB

    __bf16* hf = hout;
    __bf16* hb = hout + (size_t)EBATCH * HDIM;

    init_kernel<<<1, 256, 0, stream>>>(out, bn_sum, bn_sq);

    const int NW = 1024 * 256;  // 4H x D elements per LSTM weight matrix
    cvt_bf16_kernel<<<(NW + 255) / 256, 256, 0, stream>>>(W_ih_f, wb_ih_f, NW);
    cvt_bf16_kernel<<<(NW + 255) / 256, 256, 0, stream>>>(W_hh_f, wb_hh_f, NW);
    cvt_bf16_kernel<<<(NW + 255) / 256, 256, 0, stream>>>(W_ih_b, wb_ih_b, NW);
    cvt_bf16_kernel<<<(NW + 255) / 256, 256, 0, stream>>>(W_hh_b, wb_hh_b, NW);
    cvt_bf16_kernel<<<(256 * 512 + 255) / 256, 256, 0, stream>>>(W_lin, wb_lin, 256 * 512);

    lstm_kernel<<<256, 256, 0, stream>>>(neighbors, ent_embed,
                                         wb_ih_f, wb_hh_f, wb_ih_b, wb_hh_b,
                                         b_f, b_b, hout);

    linear_kernel<<<256, 256, 0, stream>>>(hf, hb, wb_lin, b_lin, lin);

    stats_kernel<<<32, 256, 0, stream>>>(lin, bn_sum, bn_sq);

    ctx_kernel<<<EBATCH, 256, 0, stream>>>(lin, bn_sum, bn_sq, gamma, beta, ctx);

    score_kernel<<<2 * NPAIR, 256, 0, stream>>>(trans, rel_embed, ctx,
                                                pos_h, pos_t, pos_r,
                                                neg_h, neg_t, neg_r, out);
}
